// EntropyPrunedSelfAttention_83958020702913
// MI455X (gfx1250) — compile-verified, matrix operand order checked by reading, not run on hardware
//
#include <hip/hip_runtime.h>
#include <hip/hip_bf16.h>

#define B_   4
#define N_   1024
#define C_   768
#define H_   12
#define HD_  64
#define BH_  (B_ * H_)          // 48
#define CHUNK_ 128              // keys staged per TDM transfer (16KB bf16)
#define SCALE_ 0.125f           // hd^-0.5

typedef __attribute__((ext_vector_type(16))) __bf16 v16bf;
typedef __attribute__((ext_vector_type(8)))  float  v8f;
typedef __attribute__((ext_vector_type(4)))  unsigned int u32x4;
typedef __attribute__((ext_vector_type(8)))  int i32x8;
typedef __attribute__((ext_vector_type(4)))  int i32x4;

__device__ __forceinline__ __bf16 f2bf(float f) {
  union { float f; unsigned u; } x; x.f = f;
  unsigned r = x.u + 0x7FFFu + ((x.u >> 16) & 1u);  // round-to-nearest-even
  unsigned short h = (unsigned short)(r >> 16);
  return __builtin_bit_cast(__bf16, h);
}

// K index pattern for 16-bit A/B fragments (ISA 7.12.2):
// lanes 0-15 hold K {0..7, 16..23}; lanes 16-31 hold K {8..15, 24..31}
__device__ __forceinline__ int frag_k(int v, int lane) {
  return ((v < 4) ? (2 * v) : (16 + 2 * (v - 4))) + ((lane >> 4) << 3);
}

// Fragment from a row-major [16 x K] matrix, row stride ld (elements).
// Used for A tiles (row = M) and for B tiles stored N-major (row = output col).
__device__ __forceinline__ v16bf load_frag_rowmajor(const __bf16* M, int ld, int lane) {
  v16bf f;
  const __bf16* p = M + (size_t)(lane & 15) * ld;
#pragma unroll
  for (int v = 0; v < 8; ++v) {
    int k = frag_k(v, lane);
    f[2 * v]     = p[k];
    f[2 * v + 1] = p[k + 1];
  }
  return f;
}

// B fragment from K-major storage: B[k][n] at M[k*ld + n]  (32 x 16 slice)
__device__ __forceinline__ v16bf load_frag_b_kmajor(const __bf16* M, int ld, int lane) {
  v16bf f;
  int col = lane & 15;
#pragma unroll
  for (int v = 0; v < 8; ++v) {
    int k = frag_k(v, lane);
    f[2 * v]     = M[(size_t)k * ld + col];
    f[2 * v + 1] = M[(size_t)(k + 1) * ld + col];
  }
  return f;
}

__device__ __forceinline__ v8f wmma_bf16(v16bf a, v16bf b, v8f c) {
  return __builtin_amdgcn_wmma_f32_16x16x32_bf16(false, a, false, b, (short)0, c,
                                                 false, false);
}

// 16x16 score tile: q[16x64] @ k_tile^T[64x16] via two K=32 WMMAs.
__device__ __forceinline__ v8f score_tile(v16bf qa0, v16bf qa1, const __bf16* kptr,
                                          int lane) {
  v16bf b0 = load_frag_rowmajor(kptr + 0, HD_, lane);   // keys are N-major rows
  v16bf b1 = load_frag_rowmajor(kptr + 32, HD_, lane);
  v8f s = {};
  s = wmma_bf16(qa0, b0, s);
  s = wmma_bf16(qa1, b1, s);
  return s;
}

// ---------------------------------------------------------------------------
// Tensor Data Mover: stage a 2D bf16 tile (tile_d0 elems/row x tile_d1 rows)
// from global memory into LDS.  D# layout per CDNA5 ISA ch.8 (group0/group1).
// Issued once per wave; tracked by TENSORcnt.
__device__ __forceinline__ void tdm_load_2d_bf16(unsigned lds_byte_off,
                                                 const __bf16* gptr,
                                                 unsigned tile_d0, unsigned tile_d1,
                                                 unsigned tens_d0, unsigned tens_d1,
                                                 unsigned stride0_elems) {
  unsigned long long ga = (unsigned long long)(size_t)gptr;
  u32x4 g0;
  g0[0] = 1u;                                                  // count=1, user mode
  g0[1] = lds_byte_off;                                        // lds_addr
  g0[2] = (unsigned)ga;                                        // global_addr[31:0]
  g0[3] = (unsigned)((ga >> 32) & 0x01FFFFFFu) | (2u << 30);   // ga[56:32] | type=2
  i32x8 g1;
  g1[0] = (int)(1u << 16);                                     // data_size = 2 bytes
  g1[1] = (int)((tens_d0 & 0xFFFFu) << 16);                    // tensor_dim0 lo
  g1[2] = (int)(((tens_d0 >> 16) & 0xFFFFu) |
                ((tens_d1 & 0xFFFFu) << 16));                  // dim0 hi | dim1 lo
  g1[3] = (int)(((tens_d1 >> 16) & 0xFFFFu) |
                ((tile_d0 & 0xFFFFu) << 16));                  // dim1 hi | tile_dim0
  g1[4] = (int)(tile_d1 & 0xFFFFu);                            // tile_dim1 (tile_dim2=0)
  g1[5] = (int)stride0_elems;                                  // tensor_dim0_stride lo
  g1[6] = 0;
  g1[7] = 0;
  i32x4 gz = {0, 0, 0, 0};
#if __clang_major__ >= 23
  i32x8 gz8 = {0, 0, 0, 0, 0, 0, 0, 0};
  __builtin_amdgcn_tensor_load_to_lds(g0, g1, gz, gz, gz8, 0);
#else
  __builtin_amdgcn_tensor_load_to_lds(g0, g1, gz, gz, 0);
#endif
}

// ---------------------------------------------------------------------------

__global__ void k_zero(float* p, int n) {
  int i = blockIdx.x * blockDim.x + threadIdx.x;
  if (i < n) p[i] = 0.f;
}

__global__ void k_f2bf(const float* __restrict__ src, __bf16* __restrict__ dst, int n) {
  int i = blockIdx.x * blockDim.x + threadIdx.x;
  if (i < n) dst[i] = f2bf(src[i]);
}

// QKV GEMM: x[4096x768] @ qkv_w^T[768x2304]; 4 N-tiles per wave (A-frag reuse).
// Epilogue scatters bf16 into q/k/v laid out [B,H,N,hd].
__global__ void k_qkv_gemm(const __bf16* __restrict__ xb, const __bf16* __restrict__ wb,
                           __bf16* __restrict__ qb, __bf16* __restrict__ kb,
                           __bf16* __restrict__ vb) {
  int gwid = (blockIdx.x * blockDim.x + threadIdx.x) >> 5;
  int lane = threadIdx.x & 31;
  const int NG = (3 * C_) / 64;  // 36 groups of 4 tiles
  int ng = gwid % NG, mt = gwid / NG;
  if (mt >= (B_ * N_) / 16) return;
  v8f zero = {};
  v8f acc[4];
#pragma unroll
  for (int t = 0; t < 4; ++t) acc[t] = zero;
  const __bf16* Ab = xb + (size_t)(mt * 16) * C_;
  for (int k0 = 0; k0 < C_; k0 += 32) {
    v16bf a = load_frag_rowmajor(Ab + k0, C_, lane);
#pragma unroll
    for (int t = 0; t < 4; ++t) {
      v16bf b = load_frag_rowmajor(wb + (size_t)(ng * 64 + t * 16) * C_ + k0, C_, lane);
      acc[t] = wmma_bf16(a, b, acc[t]);
    }
  }
  int col = lane & 15, rbase = (lane >> 4) << 3;
#pragma unroll
  for (int t = 0; t < 4; ++t) {
    int n = ng * 64 + t * 16 + col;
    int which = n / C_;            // 0=q 1=k 2=v (tile-uniform: 768 % 16 == 0)
    int rem = n - which * C_;
    int h = rem >> 6, d = rem & 63;
    __bf16* dst = (which == 0) ? qb : ((which == 1) ? kb : vb);
#pragma unroll
    for (int v = 0; v < 8; ++v) {
      int m = mt * 16 + v + rbase;
      int bidx = m >> 10, pos = m & (N_ - 1);
      dst[(((size_t)(bidx * H_ + h)) * N_ + pos) * HD_ + d] = f2bf(acc[t][v]);
    }
  }
}

// Pass 1: per-row online softmax (m, l). Pass 2: per-key s / plogp via atomics.
// K panel staged per 128-key chunk into LDS by the Tensor Data Mover; all 8
// waves of a block share the same (b,h).
__global__ void k_attn_stats(const __bf16* __restrict__ qb, const __bf16* __restrict__ kb,
                             float* __restrict__ mbuf, float* __restrict__ lbuf,
                             float* __restrict__ sbuf, float* __restrict__ plbuf) {
  __shared__ __bf16 kk[CHUNK_ * HD_];                 // 16 KB
  int wslot = threadIdx.x >> 5;
  int gwid = (blockIdx.x * blockDim.x + threadIdx.x) >> 5;
  int lane = threadIdx.x & 31;
  int qt = gwid & (N_ / 16 - 1);
  int bh = gwid >> 6;                                 // uniform per block
  const __bf16* q  = qb + ((size_t)bh * N_ + qt * 16) * HD_;
  const __bf16* kB = kb + (size_t)bh * N_ * HD_;
  unsigned kk_off = (unsigned)(size_t)&kk[0];
  v16bf qa0 = load_frag_rowmajor(q + 0, HD_, lane);
  v16bf qa1 = load_frag_rowmajor(q + 32, HD_, lane);

  float mrow[8], lrow[8];
#pragma unroll
  for (int v = 0; v < 8; ++v) { mrow[v] = -3.0e38f; lrow[v] = 0.f; }

  for (int ch = 0; ch < N_ / CHUNK_; ++ch) {
    __syncthreads();
    if (wslot == 0) {
      tdm_load_2d_bf16(kk_off, kB + (size_t)ch * CHUNK_ * HD_,
                       HD_, CHUNK_, HD_, CHUNK_, HD_);
      __builtin_amdgcn_s_wait_tensorcnt(0);
    }
    __syncthreads();
    for (int kt = 0; kt < CHUNK_ / 16; ++kt) {
      v8f s = score_tile(qa0, qa1, kk + kt * 16 * HD_, lane);
#pragma unroll
      for (int v = 0; v < 8; ++v) {
        float sv = s[v] * SCALE_;
        float t = sv;                       // row max over 16 cols (half-wave)
        t = fmaxf(t, __shfl_xor(t, 1, 32));
        t = fmaxf(t, __shfl_xor(t, 2, 32));
        t = fmaxf(t, __shfl_xor(t, 4, 32));
        t = fmaxf(t, __shfl_xor(t, 8, 32));
        float mn = fmaxf(mrow[v], t);
        float e = __expf(sv - mn);          // row sumexp over 16 cols
        e += __shfl_xor(e, 1, 32);
        e += __shfl_xor(e, 2, 32);
        e += __shfl_xor(e, 4, 32);
        e += __shfl_xor(e, 8, 32);
        lrow[v] = lrow[v] * __expf(mrow[v] - mn) + e;
        mrow[v] = mn;
      }
    }
  }
  int rbase = (lane >> 4) << 3;
  if ((lane & 15) == 0) {
#pragma unroll
    for (int v = 0; v < 8; ++v) {
      mbuf[(size_t)bh * N_ + qt * 16 + v + rbase] = mrow[v];
      lbuf[(size_t)bh * N_ + qt * 16 + v + rbase] = lrow[v];
    }
  }
  float logl[8];
#pragma unroll
  for (int v = 0; v < 8; ++v) logl[v] = __logf(lrow[v]);

  for (int ch = 0; ch < N_ / CHUNK_; ++ch) {
    __syncthreads();
    if (wslot == 0) {
      tdm_load_2d_bf16(kk_off, kB + (size_t)ch * CHUNK_ * HD_,
                       HD_, CHUNK_, HD_, CHUNK_, HD_);
      __builtin_amdgcn_s_wait_tensorcnt(0);
    }
    __syncthreads();
    for (int kt = 0; kt < CHUNK_ / 16; ++kt) {
      v8f s = score_tile(qa0, qa1, kk + kt * 16 * HD_, lane);
      float cs = 0.f, cp = 0.f;             // per-column partials (this half's rows)
#pragma unroll
      for (int v = 0; v < 8; ++v) {
        float lp = s[v] * SCALE_ - mrow[v] - logl[v];
        float a = __expf(lp);
        cs += a;
        cp += a * lp;
      }
      cs += __shfl_xor(cs, 16, 32);         // combine the two half-waves
      cp += __shfl_xor(cp, 16, 32);
      if (lane < 16) {
        int j = ch * CHUNK_ + kt * 16 + lane;
        atomicAdd(&sbuf[j], cs);
        atomicAdd(&plbuf[j], cp);
      }
    }
  }
}

__global__ void k_mask(const float* __restrict__ sbuf, const float* __restrict__ plbuf,
                       const int* __restrict__ cur_epoch, float* __restrict__ mask) {
  int j = blockIdx.x * blockDim.x + threadIdx.x;
  if (j >= N_) return;
  int ce = cur_epoch[0];
  float factor = 0.f;
  for (int i = 1; i <= ce; ++i) factor += 5.0f * __expf(-(float)i);
  float thr = __logf((float)C_) - factor;
  float s = sbuf[j];
  float ent = __logf(s) - plbuf[j] / s;
  mask[j] = (ent <= thr) ? 1.0f : 0.0f;
}

// Pass 3: out = (attn * mask) @ v.  K and V chunks TDM-staged into LDS; attn
// tiles round-trip through LDS (C/D layout -> A layout) and feed WMMA vs v.
__global__ void k_attn_out(const __bf16* __restrict__ qb, const __bf16* __restrict__ kb,
                           const __bf16* __restrict__ vb, const float* __restrict__ mbuf,
                           const float* __restrict__ lbuf, const float* __restrict__ mask,
                           __bf16* __restrict__ aob) {
  __shared__ __bf16 kk[CHUNK_ * HD_];                 // 16 KB
  __shared__ __bf16 vv[CHUNK_ * HD_];                 // 16 KB
  __shared__ __bf16 atile[8][16 * 32];                // 8 KB, one tile per wave
  int wslot = threadIdx.x >> 5;
  int gwid = (blockIdx.x * blockDim.x + threadIdx.x) >> 5;
  int lane = threadIdx.x & 31;
  int qt = gwid & (N_ / 16 - 1);
  int bh = gwid >> 6;                                 // uniform per block
  int b = bh / H_, h = bh % H_;
  const __bf16* q  = qb + ((size_t)bh * N_ + qt * 16) * HD_;
  const __bf16* kB = kb + (size_t)bh * N_ * HD_;
  const __bf16* vB = vb + (size_t)bh * N_ * HD_;
  unsigned kk_off = (unsigned)(size_t)&kk[0];
  unsigned vv_off = (unsigned)(size_t)&vv[0];
  v16bf qa0 = load_frag_rowmajor(q + 0, HD_, lane);
  v16bf qa1 = load_frag_rowmajor(q + 32, HD_, lane);
  int rbase = (lane >> 4) << 3;
  int col = lane & 15;

  float mrow[8], logl[8];
#pragma unroll
  for (int v = 0; v < 8; ++v) {
    mrow[v] = mbuf[(size_t)bh * N_ + qt * 16 + v + rbase];
    logl[v] = __logf(lbuf[(size_t)bh * N_ + qt * 16 + v + rbase]);
  }
  v8f zero = {};
  v8f oacc[4];
#pragma unroll
  for (int t = 0; t < 4; ++t) oacc[t] = zero;

  for (int ch = 0; ch < N_ / CHUNK_; ++ch) {
    __syncthreads();
    if (wslot == 0) {
      tdm_load_2d_bf16(kk_off, kB + (size_t)ch * CHUNK_ * HD_,
                       HD_, CHUNK_, HD_, CHUNK_, HD_);
      tdm_load_2d_bf16(vv_off, vB + (size_t)ch * CHUNK_ * HD_,
                       HD_, CHUNK_, HD_, CHUNK_, HD_);
      __builtin_amdgcn_s_wait_tensorcnt(0);
    }
    __syncthreads();
    for (int kp = 0; kp < CHUNK_ / 32; ++kp) {        // 32 keys per A tile
#pragma unroll
      for (int hlf = 0; hlf < 2; ++hlf) {
        int ktl = kp * 2 + hlf;
        int ktg = ch * (CHUNK_ / 16) + ktl;
        v8f s = score_tile(qa0, qa1, kk + (size_t)ktl * 16 * HD_, lane);
        float mk = mask[ktg * 16 + col];
#pragma unroll
        for (int v = 0; v < 8; ++v) {
          float lp = s[v] * SCALE_ - mrow[v] - logl[v];
          float a = __expf(lp) * mk;
          atile[wslot][(v + rbase) * 32 + hlf * 16 + col] = f2bf(a);
        }
      }
      __asm__ volatile("s_wait_dscnt 0" ::: "memory");  // LDS stores -> loads
      v16bf af = load_frag_rowmajor(&atile[wslot][0], 32, lane);
#pragma unroll
      for (int t = 0; t < 4; ++t) {
        v16bf bf_ = load_frag_b_kmajor(vv + (size_t)kp * 32 * HD_ + t * 16, HD_, lane);
        oacc[t] = wmma_bf16(af, bf_, oacc[t]);
      }
    }
  }
#pragma unroll
  for (int t = 0; t < 4; ++t) {
#pragma unroll
    for (int v = 0; v < 8; ++v) {
      int grow = b * N_ + qt * 16 + v + rbase;
      aob[(size_t)grow * C_ + h * HD_ + t * 16 + col] = f2bf(oacc[t][v]);
    }
  }
}

// Final projection: attn_out[4096x768] @ proj_w^T + bias -> fp32 out.
// 4 N-tiles per wave (A-frag reuse).
__global__ void k_proj_gemm(const __bf16* __restrict__ aob, const __bf16* __restrict__ pwb,
                            const float* __restrict__ pb, float* __restrict__ out) {
  int gwid = (blockIdx.x * blockDim.x + threadIdx.x) >> 5;
  int lane = threadIdx.x & 31;
  const int NG = C_ / 64;  // 12 groups of 4 tiles
  int ng = gwid % NG, mt = gwid / NG;
  if (mt >= (B_ * N_) / 16) return;
  v8f zero = {};
  v8f acc[4];
#pragma unroll
  for (int t = 0; t < 4; ++t) acc[t] = zero;
  const __bf16* Ab = aob + (size_t)(mt * 16) * C_;
  for (int k0 = 0; k0 < C_; k0 += 32) {
    v16bf a = load_frag_rowmajor(Ab + k0, C_, lane);
#pragma unroll
    for (int t = 0; t < 4; ++t) {
      v16bf b = load_frag_rowmajor(pwb + (size_t)(ng * 64 + t * 16) * C_ + k0, C_, lane);
      acc[t] = wmma_bf16(a, b, acc[t]);
    }
  }
  int col = lane & 15, rbase = (lane >> 4) << 3;
#pragma unroll
  for (int t = 0; t < 4; ++t) {
    int n = ng * 64 + t * 16 + col;
    float bias = pb[n];
#pragma unroll
    for (int v = 0; v < 8; ++v) {
      int m = mt * 16 + v + rbase;
      out[(size_t)m * C_ + n] = acc[t][v] + bias;
    }
  }
}

// ---------------------------------------------------------------------------

extern "C" void kernel_launch(void* const* d_in, const int* in_sizes, int n_in,
                              void* d_out, int out_size, void* d_ws, size_t ws_size,
                              hipStream_t stream) {
  const float* x      = (const float*)d_in[0];
  const float* qkv_w  = (const float*)d_in[1];
  const float* proj_w = (const float*)d_in[2];
  const float* proj_b = (const float*)d_in[3];
  const int*   cur_ep = (const int*)d_in[4];
  float* out = (float*)d_out;

  char* p = (char*)d_ws;
  auto take = [&](size_t bytes) -> char* {
    char* r = p;
    p += (bytes + 255) & ~(size_t)255;
    return r;
  };
  __bf16* xb   = (__bf16*)take((size_t)B_ * N_ * C_ * 2);
  __bf16* wqb  = (__bf16*)take((size_t)3 * C_ * C_ * 2);
  __bf16* wpb  = (__bf16*)take((size_t)C_ * C_ * 2);
  __bf16* qb   = (__bf16*)take((size_t)BH_ * N_ * HD_ * 2);
  __bf16* kbuf = (__bf16*)take((size_t)BH_ * N_ * HD_ * 2);
  __bf16* vbuf = (__bf16*)take((size_t)BH_ * N_ * HD_ * 2);
  float* mbuf  = (float*)take((size_t)BH_ * N_ * 4);
  float* lbuf  = (float*)take((size_t)BH_ * N_ * 4);
  float* sbuf  = (float*)take((size_t)N_ * 4);
  float* plbuf = (float*)take((size_t)N_ * 4);
  float* maskb = (float*)take((size_t)N_ * 4);
  __bf16* aob  = (__bf16*)take((size_t)B_ * N_ * C_ * 2);

  const int n1 = B_ * N_ * C_, n2 = 3 * C_ * C_, n3 = C_ * C_;
  k_f2bf<<<dim3((n1 + 255) / 256), dim3(256), 0, stream>>>(x, xb, n1);
  k_f2bf<<<dim3((n2 + 255) / 256), dim3(256), 0, stream>>>(qkv_w, wqb, n2);
  k_f2bf<<<dim3((n3 + 255) / 256), dim3(256), 0, stream>>>(proj_w, wpb, n3);
  k_zero<<<dim3(4), dim3(256), 0, stream>>>(sbuf, N_);
  k_zero<<<dim3(4), dim3(256), 0, stream>>>(plbuf, N_);

  // 256 Mtiles x 36 N-groups (4 tiles/wave), 8 waves per 256-thread block
  k_qkv_gemm<<<dim3(1152), dim3(256), 0, stream>>>(xb, wqb, qb, kbuf, vbuf);

  // 48 (b,h) x 64 q-tiles = 3072 waves; 8 waves/block, all sharing one (b,h)
  k_attn_stats<<<dim3(384), dim3(256), 0, stream>>>(qb, kbuf, mbuf, lbuf, sbuf, plbuf);

  k_mask<<<dim3(4), dim3(256), 0, stream>>>(sbuf, plbuf, cur_ep, maskb);

  k_attn_out<<<dim3(384), dim3(256), 0, stream>>>(qb, kbuf, vbuf, mbuf, lbuf, maskb, aob);

  // 256 Mtiles x 12 N-groups (4 tiles/wave)
  k_proj_gemm<<<dim3(384), dim3(256), 0, stream>>>(aob, wpb, proj_b, out);
}
